// Attention_72267119722863
// MI455X (gfx1250) — matrix operand order repaired and checked by reading, not compiled, not run on hardware
//
#include <hip/hip_runtime.h>

// ---------------------------------------------------------------------------
// Flash-attention forward (causal, GQA) for MI455X / gfx1250, wave32 WMMA.
//   Sᵀ = K·Qᵀ  (16 keys × 16 queries tiles) -> D-tile layout == A-layout of P
//   O  += P·V  (16 queries × 16 d tiles)
// f32 global data, f16 WMMA operands, f32 accumulation.
//  - double-buffered LDS, register-staged global loads, 1 barrier / 32 keys
//  - softmax in exp2 domain via raw v_exp_f32 (__builtin_amdgcn_exp2f)
//  - all f32->f16 conversion through v_cvt_pk_rtz_f16_f32
//  - V staged transposed with conflict-free ds_store_b64 key-quads
//  - Q loads / O stores non-temporal (single touch); K/V keep RT for L2 reuse
// ---------------------------------------------------------------------------

typedef _Float16 v16h __attribute__((ext_vector_type(16)));
typedef _Float16 v8h  __attribute__((ext_vector_type(8)));
typedef _Float16 v2h  __attribute__((ext_vector_type(2)));
typedef __fp16   v2fp __attribute__((ext_vector_type(2)));
typedef float    v8f  __attribute__((ext_vector_type(8)));
typedef float    v4f  __attribute__((ext_vector_type(4)));
typedef unsigned int u32;
typedef u32      v2u  __attribute__((ext_vector_type(2)));

#define NQH  16
#define NKV  4
#define HD   128
#define GQA  4          // query heads per kv head
#define SEQ  2048
#define BK   32         // keys per inner block
#define KSTR 136        // LDS row stride for K  (f16 units, padded, 16B-mult)
#define VSTR 40         // LDS row stride for V^T (f16 units, padded, 16B-mult)

union H16 { v16h v; v8h h[2]; v2h p[8]; u32 u[8]; };

__device__ __forceinline__ v2h pkrtz(float a, float b) {
  union { v2fp f; v2h h; } u;
  u.f = __builtin_amdgcn_cvt_pkrtz(a, b);   // v_cvt_pk_rtz_f16_f32
  return u.h;
}
__device__ __forceinline__ u32 pkrtz_u(float a, float b) {
  union { v2fp f; u32 u; } u;
  u.f = __builtin_amdgcn_cvt_pkrtz(a, b);
  return u.u;
}

__device__ __forceinline__ v8f wmma16(v16h a, v16h b, v8f c) {
  return __builtin_amdgcn_wmma_f32_16x16x32_f16(false, a, false, b, (short)0, c,
                                                false, false);
}

__global__ __launch_bounds__(256)
void fa_fwd_gqa_causal(const float* __restrict__ q,
                       const float* __restrict__ k,
                       const float* __restrict__ v,
                       float* __restrict__ out) {
  // double-buffered staging: ~38 KB of the 320 KB WGP LDS
  __shared__ __align__(16) _Float16 ldsK [2][BK * KSTR];  // K block, row-major
  __shared__ __align__(16) _Float16 ldsVT[2][HD * VSTR];  // V block, transposed

  const int qb    = blockIdx.x;          // 32-query block within sequence
  const int kvh   = blockIdx.y;          // kv head
  const int b     = blockIdx.z;          // sequence in packed batch
  const int tid   = threadIdx.x;
  const int lane  = tid & 31;
  const int wave  = tid >> 5;            // 8 waves
  const int g     = wave >> 1;           // query head within group
  const int halfq = wave & 1;            // which 16-query tile of the 32
  const int head  = kvh * GQA + g;
  const int q0    = qb * BK + halfq * 16;
  const int lq    = lane & 15;
  const int hi    = lane >> 4;           // lane half (K-dim split of operands)

  const int QROW = NQH * HD;             // 2048 floats per token row (q / out)
  const int KROW = NKV * HD;             // 512 floats per token row (k / v)
  // 1/sqrt(128) * log2(e): softmax evaluated as exp2(s - m)
  const float SCALE2 = 0.08838834764831845f * 1.4426950408889634f;

  // ---- Q tile -> B-operand registers (Qᵀ), scale folded in, NT loads ------
  v16h qB[4];
  {
    const float* qr = q + (size_t)(b * SEQ + q0 + lq) * QROW + head * HD;
    #pragma unroll
    for (int c = 0; c < 4; ++c) {
      const v4f* p = (const v4f*)(qr + c * 32 + hi * 16);
      v4f f[4];
      #pragma unroll
      for (int i = 0; i < 4; ++i) f[i] = __builtin_nontemporal_load(p + i);
      H16 t;
      #pragma unroll
      for (int i = 0; i < 8; ++i)
        t.p[i] = pkrtz(f[i >> 1][(i & 1) * 2]     * SCALE2,
                       f[i >> 1][(i & 1) * 2 + 1] * SCALE2);
      qB[c] = t.v;
    }
  }

  // ---- accumulators --------------------------------------------------------
  v8f o8[8];
  #pragma unroll
  for (int t = 0; t < 8; ++t)
    #pragma unroll
    for (int j = 0; j < 8; ++j) o8[t][j] = 0.0f;
  float m = -1e30f, lsum = 0.0f;

  // loader mappings (256 threads cover 32 keys x 128 d)
  const int key_ld = tid >> 3;            // K: 0..31, one key row per thread
  const int dseg   = (tid & 7) << 4;      // K: d base 0,16,...,112
  const int vkq    = (tid & 7) << 2;      // V: key quad base 0,4,...,28
  const int vdc    = (tid >> 3) << 2;     // V: d base 0,4,...,124
  const size_t seqbase = (size_t)(b * SEQ) * KROW + kvh * HD;

  float kf[16];      // staged K row segment
  float vf[4][4];    // staged V 4keys x 4d patch

  // ---- helpers as lambdas --------------------------------------------------
  auto load_block = [&](int blk) {
    const float* kp = k + seqbase + (size_t)(blk * BK + key_ld) * KROW + dseg;
    #pragma unroll
    for (int i = 0; i < 4; ++i) {
      v4f f = *((const v4f*)kp + i);
      kf[4*i+0] = f[0]; kf[4*i+1] = f[1]; kf[4*i+2] = f[2]; kf[4*i+3] = f[3];
    }
    #pragma unroll
    for (int i = 0; i < 4; ++i) {
      const v4f f = *(const v4f*)(v + seqbase +
                                  (size_t)(blk * BK + vkq + i) * KROW + vdc);
      vf[i][0] = f[0]; vf[i][1] = f[1]; vf[i][2] = f[2]; vf[i][3] = f[3];
    }
  };
  auto store_block = [&](int buf) {
    H16 hk;
    #pragma unroll
    for (int i = 0; i < 8; ++i)
      hk.p[i] = pkrtz(kf[2*i], kf[2*i+1]);
    *(v8h*)&ldsK[buf][key_ld * KSTR + dseg]     = hk.h[0];
    *(v8h*)&ldsK[buf][key_ld * KSTR + dseg + 8] = hk.h[1];
    #pragma unroll
    for (int dd = 0; dd < 4; ++dd) {   // key-quad packed b64 stores, 0-conflict
      v2u st;
      st[0] = pkrtz_u(vf[0][dd], vf[1][dd]);
      st[1] = pkrtz_u(vf[2][dd], vf[3][dd]);
      *(v2u*)&ldsVT[buf][(vdc + dd) * VSTR + vkq] = st;
    }
  };

  // ---- prologue: stage block 0 into buffer 0 -------------------------------
  load_block(0);
  store_block(0);
  __syncthreads();

  for (int kb = 0; kb <= qb; ++kb) {
    const int  cur      = kb & 1;
    const bool havenext = kb < qb;

    // issue next block's global loads NOW; compute below hides the latency
    if (havenext) load_block(kb + 1);

    const _Float16* Kc = &ldsK [cur][0];
    const _Float16* Vc = &ldsVT[cur][0];

    // ---- Sᵀ = K·Qᵀ : two 16(key)x16(query) tiles, contract D=128 ----------
    v8f st0 = {}, st1 = {};
    #pragma unroll
    for (int c = 0; c < 4; ++c) {
      H16 a0, a1;
      const _Float16* p0 = Kc + lq * KSTR + c * 32 + hi * 16;
      a0.h[0] = *(const v8h*)p0;
      a0.h[1] = *(const v8h*)(p0 + 8);
      const _Float16* p1 = Kc + (16 + lq) * KSTR + c * 32 + hi * 16;
      a1.h[0] = *(const v8h*)p1;
      a1.h[1] = *(const v8h*)(p1 + 8);
      st0 = wmma16(a0.v, qB[c], st0);
      st1 = wmma16(a1.v, qB[c], st1);
    }

    // ---- causal mask (only the diagonal block needs it) --------------------
    if (kb == qb) {
      const int qg  = q0 + lq;
      const int k0g = kb * BK + hi * 8;   // D-tile: VGPR j -> key k0g+j
      #pragma unroll
      for (int j = 0; j < 8; ++j) {
        if (k0g + j      > qg) st0[j] = -1e30f;
        if (k0g + j + 16 > qg) st1[j] = -1e30f;
      }
    }

    // ---- online softmax (row of query lq split across lanes lq, lq+16) ----
    float bm = -1e30f;
    #pragma unroll
    for (int j = 0; j < 8; ++j) { bm = fmaxf(bm, st0[j]); bm = fmaxf(bm, st1[j]); }
    bm = fmaxf(bm, __shfl_xor(bm, 16, 32));
    const float mnew  = fmaxf(m, bm);
    const float alpha = __builtin_amdgcn_exp2f(m - mnew);  // raw v_exp_f32
    H16 pa;                  // == A-operand layout for P·V, no shuffles needed
    float rs = 0.0f;
    #pragma unroll
    for (int jj = 0; jj < 4; ++jj) {
      float a0 = __builtin_amdgcn_exp2f(st0[2*jj]     - mnew);
      float a1 = __builtin_amdgcn_exp2f(st0[2*jj + 1] - mnew);
      float b0 = __builtin_amdgcn_exp2f(st1[2*jj]     - mnew);
      float b1 = __builtin_amdgcn_exp2f(st1[2*jj + 1] - mnew);
      rs += (a0 + a1) + (b0 + b1);
      pa.p[jj]     = pkrtz(a0, a1);
      pa.p[4 + jj] = pkrtz(b0, b1);
    }
    rs += __shfl_xor(rs, 16, 32);
    lsum = lsum * alpha + rs;
    m = mnew;

    // rescale accumulators: O-tile VGPR j holds query (hi*8+j)
    #pragma unroll
    for (int j = 0; j < 8; ++j) {
      const float aj = __shfl(alpha, hi * 8 + j, 32);
      #pragma unroll
      for (int t = 0; t < 8; ++t) o8[t][j] *= aj;
    }

    // ---- O += P·V : 8 d-tiles, contract 32 keys ---------------------------
    #pragma unroll
    for (int t = 0; t < 8; ++t) {
      H16 vb;
      const _Float16* vp = Vc + (t * 16 + lq) * VSTR + hi * 8;
      vb.h[0] = *(const v8h*)vp;
      vb.h[1] = *(const v8h*)(vp + 16);
      o8[t] = wmma16(pa.v, vb.v, o8[t]);
    }

    // ---- write staged next block into the other buffer --------------------
    if (havenext) store_block(cur ^ 1);
    __syncthreads();   // next buffer ready; current buffer free for overwrite
  }

  // ---- normalize and store (non-temporal, single touch) --------------------
  float linv[8];
  #pragma unroll
  for (int j = 0; j < 8; ++j) {
    const float lj = __shfl(lsum, hi * 8 + j, 32);
    linv[j] = __builtin_amdgcn_rcpf(lj);   // diagonal guarantees lj > 0
  }
  float* ob = out + (size_t)(b * SEQ + q0 + hi * 8) * QROW + head * HD + lq;
  #pragma unroll
  for (int t = 0; t < 8; ++t)
    #pragma unroll
    for (int j = 0; j < 8; ++j)
      __builtin_nontemporal_store(o8[t][j] * linv[j],
                                  ob + (size_t)j * QROW + t * 16);
}

extern "C" void kernel_launch(void* const* d_in, const int* in_sizes, int n_in,
                              void* d_out, int out_size, void* d_ws, size_t ws_size,
                              hipStream_t stream) {
  (void)n_in; (void)d_ws; (void)ws_size; (void)out_size;
  const float* q = (const float*)d_in[0];
  const float* k = (const float*)d_in[1];
  const float* v = (const float*)d_in[2];
  float* out = (float*)d_out;

  const int T = in_sizes[0] / (NQH * HD);   // packed tokens
  const int B = T / SEQ;                    // sequences (reference: 4 x 2048)
  dim3 grid(SEQ / BK, NKV, B);
  dim3 block(256);
  fa_fwd_gqa_causal<<<grid, block, 0, stream>>>(q, k, v, out);
}
